// VanillaSelfAttention_36919538876797
// MI455X (gfx1250) — matrix operand order repaired and checked by reading, not compiled, not run on hardware
//
#include <hip/hip_runtime.h>

typedef __attribute__((ext_vector_type(16))) _Float16 v16h;
typedef __attribute__((ext_vector_type(8)))  _Float16 v8h;
typedef __attribute__((ext_vector_type(8)))  float    v8f;

#define RTOT 20000   // B*N
#define RPAD 20032   // 313 * 64, padded row count for guard-free WMMA tiles
#define CIN  128
#define NB   2
#define NN   10000
#define NHH  4
#define NPP  20
#define DHH  32
#define HW_  100
#define CCAT 368     // 128 (val) + 160 (off) + 80 (attn)

union V16U { v16h v; v8h h[2]; };

// A operand (16x32 f16, M=lane%16): lanes 0-15 hold K=0..15, lanes 16-31 hold K=16..31 (contiguous)
__device__ __forceinline__ v16h load_a(const _Float16* __restrict__ row, int kb, int hl) {
  return *(const v16h*)(row + kb * 32 + hl * 16);
}
// B operand (32x16 f16, N=lane%16): lanes 0-15 hold K=0..15, lanes 16-31 hold K=16..31 (contiguous)
__device__ __forceinline__ v16h load_b(const _Float16* __restrict__ col, int kb, int hl) {
  return *(const v16h*)(col + kb * 32 + hl * 16);
}

// Y[RPAD x ldy] = X[RPAD x 128](f16) @ W[Cout x 128]^T(f16) + bias (+ resid)
// MODE 0: f32 out, unguarded (padded buffers)
// MODE 1: f16 out, unguarded (padded buffers)
// MODE 2: f32 out + residual, guarded only in the wave-uniform tail
template <int MODE>
__global__ __launch_bounds__(256) void gemm_wmma_kernel(
    const _Float16* __restrict__ X, const _Float16* __restrict__ W,
    const float* __restrict__ bias, void* __restrict__ Yout,
    const float* __restrict__ resid, int R, int Cout, int ldy)
{
  const int lane = threadIdx.x & 31;
  const int wave = threadIdx.x >> 5;
  const int nMG  = RPAD >> 6;                       // 313 row groups of 64
  const int g    = blockIdx.x * (blockDim.x >> 5) + wave;
  const int tot  = nMG * (Cout >> 4);
  if (g >= tot) return;
  const int mg = g % nMG;
  const int ng = g / nMG;
  const int row_base = mg << 6;
  const int col_base = ng << 4;
  const int l16 = lane & 15;
  const int hl  = lane >> 4;

  const _Float16* wrow = W + (size_t)(col_base + l16) * CIN;
  v16h b0 = load_b(wrow, 0, hl);
  v16h b1 = load_b(wrow, 1, hl);
  v16h b2 = load_b(wrow, 2, hl);
  v16h b3 = load_b(wrow, 3, hl);
  const float bv = bias[col_base + l16];

#pragma unroll
  for (int m = 0; m < 4; ++m) {
    const _Float16* xrow = X + (size_t)(row_base + m * 16 + l16) * CIN;
    v16h a0 = load_a(xrow, 0, hl);
    v16h a1 = load_a(xrow, 1, hl);
    v16h a2 = load_a(xrow, 2, hl);
    v16h a3 = load_a(xrow, 3, hl);
    v8f c = {bv, bv, bv, bv, bv, bv, bv, bv};
    c = __builtin_amdgcn_wmma_f32_16x16x32_f16(false, a0, false, b0, (short)0, c, false, false);
    c = __builtin_amdgcn_wmma_f32_16x16x32_f16(false, a1, false, b1, (short)0, c, false, false);
    c = __builtin_amdgcn_wmma_f32_16x16x32_f16(false, a2, false, b2, (short)0, c, false, false);
    c = __builtin_amdgcn_wmma_f32_16x16x32_f16(false, a3, false, b3, (short)0, c, false, false);

    const int mrow0 = row_base + m * 16 + (hl << 3);  // D: lane l, elem r -> M = r + 8*(l/16)
    const size_t o0 = (size_t)mrow0 * ldy + col_base + l16;

    if (MODE == 0) {
      float* Y = (float*)Yout;
#pragma unroll
      for (int r = 0; r < 8; ++r) Y[o0 + (size_t)r * ldy] = c[r];
    } else if (MODE == 1) {
      _Float16* Y = (_Float16*)Yout;
#pragma unroll
      for (int r = 0; r < 8; ++r) Y[o0 + (size_t)r * ldy] = (_Float16)c[r];
    } else {
      float* Y = (float*)Yout;
      if (row_base + 64 <= R) {   // wave-uniform fast path: batch residual loads, then stores
        float rv[8];
#pragma unroll
        for (int r = 0; r < 8; ++r) rv[r] = resid[o0 + (size_t)r * ldy];
#pragma unroll
        for (int r = 0; r < 8; ++r) Y[o0 + (size_t)r * ldy] = c[r] + rv[r];
      } else {                    // only the final row group ever takes this
#pragma unroll
        for (int r = 0; r < 8; ++r) {
          if (mrow0 + r < R) {
            size_t o = o0 + (size_t)r * ldy;
            Y[o] = c[r] + resid[o];
          }
        }
      }
    }
  }
}

__device__ __forceinline__ float corner_val(const float* __restrict__ vbase,
                                            int ix, int iy, float w) {
  bool valid = (ix >= 0) && (ix < HW_) && (iy >= 0) && (iy < HW_);
  int cx = ix < 0 ? 0 : (ix > HW_ - 1 ? HW_ - 1 : ix);
  int cy = iy < 0 ? 0 : (iy > HW_ - 1 ? HW_ - 1 : iy);
  float v = vbase[(size_t)(cy * HW_ + cx) * CCAT];
  return valid ? w * v : 0.0f;
}

// one wave per (b,n,h); lane = channel d within head
__global__ __launch_bounds__(256) void sample_kernel(
    const float* __restrict__ Yproj, _Float16* __restrict__ S)
{
  const int lane = threadIdx.x & 31;
  const int wave = threadIdx.x >> 5;
  int g = blockIdx.x * (blockDim.x >> 5) + wave;
  if (g >= RTOT * NHH) return;
  const int h  = g & 3;
  const int bn = g >> 2;
  const int b  = bn / NN;
  const int n  = bn % NN;
  const float* yrow = Yproj + (size_t)bn * CCAT;

  // softmax over the 20 attn logits (redundant per lane; broadcast reads)
  float logits[NPP];
  float mx = -1e30f;
#pragma unroll
  for (int p = 0; p < NPP; ++p) {
    logits[p] = yrow[288 + h * NPP + p];
    mx = fmaxf(mx, logits[p]);
  }
  float ssum = 0.0f;
#pragma unroll
  for (int p = 0; p < NPP; ++p) { logits[p] = __expf(logits[p] - mx); ssum += logits[p]; }
  const float inv = 1.0f / ssum;

  const int iz  = n / HW_;
  const int ixn = n % HW_;
  const float* vbase = Yproj + (size_t)b * NN * CCAT + h * DHH + lane;
  float acc = 0.0f;
#pragma unroll
  for (int p = 0; p < NPP; ++p) {
    float o0 = yrow[128 + h * NPP * 2 + p * 2 + 0];
    float o1 = yrow[128 + h * NPP * 2 + p * 2 + 1];
    // x = loc0*HW-0.5 = iz + off0 ; y = loc1*HW-0.5 = ixn + off1
    float x = (float)iz + o0;
    float y = (float)ixn + o1;
    float x0f = floorf(x), y0f = floorf(y);
    int ix0 = (int)x0f, iy0 = (int)y0f;
    int ix1 = ix0 + 1, iy1 = iy0 + 1;
    float wx1 = x - x0f, wy1 = y - y0f;
    float wx0 = 1.0f - wx1, wy0 = 1.0f - wy1;
    float aw = logits[p] * inv;
    float s = corner_val(vbase, ix0, iy0, wx0 * wy0)
            + corner_val(vbase, ix1, iy0, wx1 * wy0)
            + corner_val(vbase, ix0, iy1, wx0 * wy1)
            + corner_val(vbase, ix1, iy1, wx1 * wy1);
    acc += aw * s;
  }
  S[(size_t)bn * CIN + h * DHH + lane] = (_Float16)acc;
}

__global__ void k_addcvt(const float* __restrict__ a, const float* __restrict__ b,
                         _Float16* __restrict__ o, int n) {
  int i = blockIdx.x * blockDim.x + threadIdx.x;
  if (i < n) o[i] = (_Float16)(a[i] + b[i]);
}
__global__ void k_cvt(const float* __restrict__ a, _Float16* __restrict__ o, int n) {
  int i = blockIdx.x * blockDim.x + threadIdx.x;
  if (i < n) o[i] = (_Float16)a[i];
}
__global__ void k_bias_cat(const float* __restrict__ bv, const float* __restrict__ bo,
                           const float* __restrict__ ba, float* __restrict__ o) {
  int i = blockIdx.x * blockDim.x + threadIdx.x;
  if (i < 128)      o[i] = bv[i];
  else if (i < 288) o[i] = bo[i - 128];
  else if (i < 368) o[i] = ba[i - 288];
}

extern "C" void kernel_launch(void* const* d_in, const int* in_sizes, int n_in,
                              void* d_out, int out_size, void* d_ws, size_t ws_size,
                              hipStream_t stream) {
  const float* query  = (const float*)d_in[0];
  const float* qpos   = (const float*)d_in[1];
  const float* W_val  = (const float*)d_in[2];
  const float* b_val  = (const float*)d_in[3];
  const float* W_off  = (const float*)d_in[4];
  const float* b_off  = (const float*)d_in[5];
  const float* W_attn = (const float*)d_in[6];
  const float* b_attn = (const float*)d_in[7];
  const float* W_o1   = (const float*)d_in[8];
  const float* b_o1   = (const float*)d_in[9];
  const float* W_o2   = (const float*)d_in[10];
  const float* b_o2   = (const float*)d_in[11];

  char* ws = (char*)d_ws;
  size_t off = 0;
  auto walloc = [&](size_t bytes) -> void* {
    void* p = ws + off;
    off = (off + bytes + 255) & ~(size_t)255;
    return p;
  };
  _Float16* qf16   = (_Float16*)walloc((size_t)RPAD * CIN * 2);   // reused as Y1 later
  _Float16* Wcat   = (_Float16*)walloc((size_t)CCAT * CIN * 2);
  _Float16* Wo1h   = (_Float16*)walloc((size_t)CIN * CIN * 2);
  _Float16* Wo2h   = (_Float16*)walloc((size_t)CIN * CIN * 2);
  float*    bcat   = (float*)   walloc((size_t)CCAT * 4);
  float*    Yproj  = (float*)   walloc((size_t)RPAD * CCAT * 4);
  _Float16* Sf16   = (_Float16*)walloc((size_t)RPAD * CIN * 2);
  _Float16* Y1h    = qf16;  // alias: qf16 dead after proj GEMM; same padded size

  // 1) q = query + query_pos -> f16
  {
    int n = RTOT * CIN;
    k_addcvt<<<(n + 255) / 256, 256, 0, stream>>>(query, qpos, qf16, n);
  }
  // 2) weight conversions (Wcat = [W_val; W_off; W_attn])
  k_cvt<<<(128 * 128 + 255) / 256, 256, 0, stream>>>(W_val,  Wcat,             128 * 128);
  k_cvt<<<(160 * 128 + 255) / 256, 256, 0, stream>>>(W_off,  Wcat + 128 * 128, 160 * 128);
  k_cvt<<<( 80 * 128 + 255) / 256, 256, 0, stream>>>(W_attn, Wcat + 288 * 128,  80 * 128);
  k_cvt<<<(128 * 128 + 255) / 256, 256, 0, stream>>>(W_o1, Wo1h, 128 * 128);
  k_cvt<<<(128 * 128 + 255) / 256, 256, 0, stream>>>(W_o2, Wo2h, 128 * 128);
  k_bias_cat<<<2, 256, 0, stream>>>(b_val, b_off, b_attn, bcat);

  const int nMG = RPAD / 64;  // 313
  // 3) fused projection GEMM: Yproj[RPAD x 368] = q @ [Wv;Wo;Wa]^T + bcat
  {
    int tot = nMG * (CCAT / 16);
    gemm_wmma_kernel<0><<<(tot + 7) / 8, 256, 0, stream>>>(
        qf16, Wcat, bcat, (void*)Yproj, (const float*)nullptr, RTOT, CCAT, CCAT);
  }
  // 4) softmax + bilinear sampling -> Sf16[R x 128]
  {
    int groups = RTOT * NHH;
    sample_kernel<<<(groups + 7) / 8, 256, 0, stream>>>(Yproj, Sf16);
  }
  // 5) out1 GEMM: Y1h = S @ W_out1^T + b_out1  (f16 out)
  {
    int tot = nMG * (CIN / 16);
    gemm_wmma_kernel<1><<<(tot + 7) / 8, 256, 0, stream>>>(
        Sf16, Wo1h, b_o1, (void*)Y1h, (const float*)nullptr, RTOT, CIN, CIN);
  }
  // 6) out2 GEMM + residual: d_out = Y1 @ W_out2^T + b_out2 + query
  {
    int tot = nMG * (CIN / 16);
    gemm_wmma_kernel<2><<<(tot + 7) / 8, 256, 0, stream>>>(
        Y1h, Wo2h, b_o2, d_out, query, RTOT, CIN, CIN);
  }
}
